// GATReg_46883863003260
// MI455X (gfx1250) — matrix operand order repaired and checked by reading, not compiled, not run on hardware
//
#include <hip/hip_runtime.h>
#include <hip/hip_bf16.h>

// ---------------------------------------------------------------------------
// GATv2Conv (H=4, C=16, D_IN=128) + ReLU + linear head for MI455X (gfx1250).
//  - Projections x@W_l / x@W_r via v_wmma_f32_16x16x32_bf16 with bf16
//    split-precision (hi/lo) for ~fp32 accuracy.
//  - Edge passes are L2-resident gathers (x_l/x_r are 25.6 MB each << 192 MB
//    L2) with f32/u32 atomics for segment max / sum / aggregate.
// ---------------------------------------------------------------------------

#define HEADS 4
#define CDIM 16
#define HC 64
#define DIN 128
#define NEG_SLOPE 0.2f

typedef __attribute__((ext_vector_type(16))) __bf16 v16bf;
typedef __attribute__((ext_vector_type(8)))  float  v8f;

// ---- bf16 helpers (bit-level, no reliance on __bf16 arithmetic) -----------
static __device__ __forceinline__ __bf16 f32_to_bf16(float f) {
    unsigned u = __float_as_uint(f);
    unsigned r = (u + 0x7FFFu + ((u >> 16) & 1u)) >> 16;   // round-nearest-even
    unsigned short s = (unsigned short)r;
    __bf16 b;
    __builtin_memcpy(&b, &s, 2);
    return b;
}
static __device__ __forceinline__ float bf16_to_f32(__bf16 b) {
    unsigned short s;
    __builtin_memcpy(&s, &b, 2);
    return __uint_as_float(((unsigned)s) << 16);
}

// Order-preserving float -> uint encoding for atomicMax-based segment max.
static __device__ __forceinline__ unsigned enc_f32(float f) {
    unsigned u = __float_as_uint(f);
    return (u & 0x80000000u) ? ~u : (u | 0x80000000u);
}
static __device__ __forceinline__ float dec_u32(unsigned u) {
    unsigned b = (u & 0x80000000u) ? (u & 0x7FFFFFFFu) : ~u;
    return __uint_as_float(b);
}

// ---------------------------------------------------------------------------
// Kernel 1: projection GEMM  out[N,64] = x[N,128] @ W[128,64] + bvec
// One wave per 16x16 output tile; K=128 consumed in four 16x16x32 WMMA steps,
// each done as 3 bf16 WMMAs (hi*hi + hi*lo + lo*hi) for fp32-class accuracy.
// ---------------------------------------------------------------------------
__global__ void proj_wmma_kernel(const float* __restrict__ x,
                                 const float* __restrict__ W,
                                 const float* __restrict__ bvec,
                                 float* __restrict__ out, int N) {
    const int lane = threadIdx.x & 31;
    const int wave = threadIdx.x >> 5;
    const int rowTiles = (N + 15) >> 4;
    const long tile = (long)blockIdx.x * (blockDim.x >> 5) + wave;
    if (tile >= (long)rowTiles * 4) return;          // wave-uniform exit

    const int colTile = (int)(tile & 3);
    const int rowTile = (int)(tile >> 2);
    const int m   = lane & 15;                        // M (A) / N (B,C) index
    const int hiL = lane >> 4;                        // lane group 0/1

    int row = rowTile * 16 + m;
    if (row >= N) row = N - 1;                        // clamp: keep EXEC full
    const float* __restrict__ xrow = x + (size_t)row * DIN;
    const int col = colTile * 16 + m;

    v8f acc = {0.f, 0.f, 0.f, 0.f, 0.f, 0.f, 0.f, 0.f};

    #pragma unroll
    for (int kc = 0; kc < 4; ++kc) {
        const int K0 = kc * 32;
        v16bf a_hi, a_lo, b_hi, b_lo;

        // A fragment (16x32 bf16): VGPR p holds K pair; lanes 0-15 cover
        // K={0..7,16..23}, lanes 16-31 cover K={8..15,24..31}.
        #pragma unroll
        for (int p = 0; p < 8; ++p) {
            const int kbase = (p < 4) ? (2 * p + (hiL ? 8 : 0))
                                      : (2 * (p - 4) + 16 + (hiL ? 8 : 0));
            const float f0 = xrow[K0 + kbase];
            const float f1 = xrow[K0 + kbase + 1];
            const __bf16 h0 = f32_to_bf16(f0);
            const __bf16 h1 = f32_to_bf16(f1);
            a_hi[2 * p]     = h0;
            a_hi[2 * p + 1] = h1;
            a_lo[2 * p]     = f32_to_bf16(f0 - bf16_to_f32(h0));
            a_lo[2 * p + 1] = f32_to_bf16(f1 - bf16_to_f32(h1));
        }

        // B fragment (32x16 bf16): N = lane%16, element i -> K = 8*hiL + i (i<8), 16 + 8*hiL + (i-8) (i>=8).
        #pragma unroll
        for (int i = 0; i < 16; ++i) {
            const int k = K0 + ((i < 8) ? i : i + 8) + (hiL ? 8 : 0);
            const float f = W[(size_t)k * HC + col];
            const __bf16 h = f32_to_bf16(f);
            b_hi[i] = h;
            b_lo[i] = f32_to_bf16(f - bf16_to_f32(h));
        }

        // Split-precision accumulate: hi*hi + hi*lo + lo*hi.
        acc = __builtin_amdgcn_wmma_f32_16x16x32_bf16(
                  false, a_hi, false, b_hi, (short)0, acc, false, false);
        acc = __builtin_amdgcn_wmma_f32_16x16x32_bf16(
                  false, a_hi, false, b_lo, (short)0, acc, false, false);
        acc = __builtin_amdgcn_wmma_f32_16x16x32_bf16(
                  false, a_lo, false, b_hi, (short)0, acc, false, false);
    }

    const float bias = bvec[col];
    #pragma unroll
    for (int r = 0; r < 8; ++r) {
        const int mrow = rowTile * 16 + r + (hiL ? 8 : 0);
        if (mrow < N) out[(size_t)mrow * HC + col] = acc[r] + bias;
    }
}

// ---------------------------------------------------------------------------
// Kernel 2: per-(edge, head) attention logits + segment max (encoded u32).
// ---------------------------------------------------------------------------
__global__ void edge_logits_kernel(const int* __restrict__ ei, int E, int EP,
                                   const float* __restrict__ x_l,
                                   const float* __restrict__ x_r,
                                   const float* __restrict__ att,
                                   float* __restrict__ logits,
                                   unsigned* __restrict__ segmax) {
    const long idx = (long)blockIdx.x * blockDim.x + threadIdx.x;
    if (idx >= (long)EP * HEADS) return;
    const long e = idx >> 2;
    const int  h = (int)(idx & 3);
    int s, d;
    if (e < E) { s = ei[e]; d = ei[(long)E + e]; }
    else       { s = d = (int)(e - E); }

    const float* __restrict__ pl = x_l + (size_t)s * HC + h * CDIM;
    const float* __restrict__ pr = x_r + (size_t)d * HC + h * CDIM;
    const float* __restrict__ pa = att + h * CDIM;

    float acc = 0.f;
    #pragma unroll
    for (int c = 0; c < CDIM; ++c) {
        float v = pl[c] + pr[c];
        v = (v >= 0.f) ? v : NEG_SLOPE * v;
        acc += v * pa[c];
    }
    logits[idx] = acc;
    atomicMax(segmax + (size_t)d * HEADS + h, enc_f32(acc));
}

// ---------------------------------------------------------------------------
// Kernel 3: expw = exp(logit - segmax[dst]); segment sum.
// ---------------------------------------------------------------------------
__global__ void edge_expsum_kernel(const int* __restrict__ ei, int E, int EP,
                                   float* __restrict__ logits,  // in: logit, out: expw
                                   const unsigned* __restrict__ segmax,
                                   float* __restrict__ segsum) {
    const long idx = (long)blockIdx.x * blockDim.x + threadIdx.x;
    if (idx >= (long)EP * HEADS) return;
    const long e = idx >> 2;
    const int  h = (int)(idx & 3);
    int d;
    if (e < E) d = ei[(long)E + e];
    else       d = (int)(e - E);

    const float mx = dec_u32(segmax[(size_t)d * HEADS + h]);
    const float w  = __expf(logits[idx] - mx);
    logits[idx] = w;
    atomicAdd(segsum + (size_t)d * HEADS + h, w);
}

// ---------------------------------------------------------------------------
// Kernel 4: scatter messages  agg[dst] += (expw/segsum[dst]) * x_l[src].
// ---------------------------------------------------------------------------
__global__ void edge_scatter_kernel(const int* __restrict__ ei, int E, int EP,
                                    const float* __restrict__ expw,
                                    const float* __restrict__ segsum,
                                    const float* __restrict__ x_l,
                                    float* __restrict__ agg) {
    const long idx = (long)blockIdx.x * blockDim.x + threadIdx.x;
    if (idx >= (long)EP * HEADS) return;
    const long e = idx >> 2;
    const int  h = (int)(idx & 3);
    int s, d;
    if (e < E) { s = ei[e]; d = ei[(long)E + e]; }
    else       { s = d = (int)(e - E); }

    const float alpha = expw[idx] / segsum[(size_t)d * HEADS + h];
    const float* __restrict__ pl = x_l + (size_t)s * HC + h * CDIM;
    float* __restrict__ po = agg + (size_t)d * HC + h * CDIM;
    #pragma unroll
    for (int c = 0; c < CDIM; ++c)
        atomicAdd(po + c, alpha * pl[c]);
}

// ---------------------------------------------------------------------------
// Kernel 5: out[n] = W_head . relu(agg[n] + bias) + b_head
// ---------------------------------------------------------------------------
__global__ void head_kernel(const float* __restrict__ agg,
                            const float* __restrict__ bias,
                            const float* __restrict__ W_head,
                            const float* __restrict__ b_head,
                            float* __restrict__ out, int N) {
    const long n = (long)blockIdx.x * blockDim.x + threadIdx.x;
    if (n >= N) return;
    const float* __restrict__ pa = agg + (size_t)n * HC;
    float acc = b_head[0];
    #pragma unroll
    for (int j = 0; j < HC; ++j) {
        float v = pa[j] + bias[j];
        v = (v > 0.f) ? v : 0.f;
        acc += v * W_head[j];
    }
    out[n] = acc;
}

// ---------------------------------------------------------------------------
extern "C" void kernel_launch(void* const* d_in, const int* in_sizes, int n_in,
                              void* d_out, int out_size, void* d_ws, size_t ws_size,
                              hipStream_t stream) {
    const float* x      = (const float*)d_in[0];
    const int*   ei     = (const int*)  d_in[1];
    const float* W_l    = (const float*)d_in[2];
    const float* b_l    = (const float*)d_in[3];
    const float* W_r    = (const float*)d_in[4];
    const float* b_r    = (const float*)d_in[5];
    const float* att    = (const float*)d_in[6];
    const float* bias   = (const float*)d_in[7];
    const float* W_head = (const float*)d_in[8];
    const float* b_head = (const float*)d_in[9];
    float* out = (float*)d_out;

    const int N  = in_sizes[0] / DIN;
    const int E  = in_sizes[1] / 2;
    const int EP = E + N;          // edges + self loops

    // Workspace layout (floats): x_l | x_r | logits/expw | segmax | segsum | agg
    float*    x_l    = (float*)d_ws;
    float*    x_r    = x_l + (size_t)N * HC;
    float*    logits = x_r + (size_t)N * HC;
    unsigned* segmax = (unsigned*)(logits + (size_t)EP * HEADS);
    float*    segsum = (float*)(segmax + (size_t)N * HEADS);
    float*    agg    = segsum + (size_t)N * HEADS;

    // Zero the accumulator region every call (capture-safe async memset).
    // segmax encoded-0 acts as -inf identity; every node has a self loop.
    hipMemsetAsync(segmax, 0,
                   ((size_t)N * HEADS * 2 + (size_t)N * HC) * sizeof(float),
                   stream);

    // Projections via WMMA.
    {
        const int rowTiles = (N + 15) / 16;
        const int tiles    = rowTiles * 4;          // 4 column tiles of 16
        const int wavesPerBlock = 8;                // 256 threads
        const int blocks = (tiles + wavesPerBlock - 1) / wavesPerBlock;
        proj_wmma_kernel<<<blocks, 256, 0, stream>>>(x, W_l, b_l, x_l, N);
        proj_wmma_kernel<<<blocks, 256, 0, stream>>>(x, W_r, b_r, x_r, N);
    }

    // Edge passes.
    {
        const long nthr = (long)EP * HEADS;
        const int blocks = (int)((nthr + 255) / 256);
        edge_logits_kernel<<<blocks, 256, 0, stream>>>(ei, E, EP, x_l, x_r, att,
                                                       logits, segmax);
        edge_expsum_kernel<<<blocks, 256, 0, stream>>>(ei, E, EP, logits,
                                                       segmax, segsum);
        edge_scatter_kernel<<<blocks, 256, 0, stream>>>(ei, E, EP, logits,
                                                        segsum, x_l, agg);
    }

    // Head.
    head_kernel<<<(N + 255) / 256, 256, 0, stream>>>(agg, bias, W_head, b_head,
                                                     out, N);
}
